// AttentionModel_69733089017953
// MI455X (gfx1250) — matrix operand order repaired and checked by reading, not compiled, not run on hardware
//
#include <hip/hip_runtime.h>
#include <math.h>

// MI455X / gfx1250, wave32. Single-pass (flash-style) attention-context:
//   energy -> online softmax -> weighted mean, one HBM read of the 512MB tensor
//   (roofline floor ~22us at 23.3 TB/s; a two-pass version would be ~44us).
// Energy dot-products run on the matrix pipe (v_wmma_f32_16x16x32_bf16) with
// bf16 hi/lo operand splitting for ~fp32 accuracy; softmax + context on VALU.
// Tile t+1 is prefetched (global_prefetch_b8) while tile t is computed.

typedef __attribute__((ext_vector_type(16))) __bf16 v16bf;
typedef __attribute__((ext_vector_type(8)))  float  v8f;

#define B_DIM   64
#define L_DIM   4096
#define D_DIM   512
#define NCHUNK  16
#define CH_ROWS (L_DIM / NCHUNK)    // 256 rows per block
#define T_ROWS  64                  // rows per LDS tile (16 per wave, 4 waves)
#define N_TILE  (CH_ROWS / T_ROWS)  // 4
#define ROWPAD  516                 // 512 + 4 floats pad -> bank-conflict-free
#define PARTSZ  514                 // 512 ctx + running max + running sum

__global__ __launch_bounds__(128) void attn_pass1(
    const float* __restrict__ hidden,   // [B, D]
    const float* __restrict__ enc,      // [B, L, D]
    float* __restrict__ part)           // [B*NCHUNK, PARTSZ]
{
    __shared__ float s_enc[T_ROWS * ROWPAD];                 // 132096 B
    __shared__ __attribute__((aligned(32))) __bf16 s_hh[D_DIM];
    __shared__ __attribute__((aligned(32))) __bf16 s_hl[D_DIM];

    const int chunk = blockIdx.x;
    const int b     = blockIdx.y;
    const int tid   = threadIdx.x;          // 0..127
    const int wave  = tid >> 5;             // 0..3
    const int lane  = tid & 31;
    const int half  = lane >> 4;            // lane group 0-15 vs 16-31
    const int mrow  = lane & 15;

    // Stage hidden[b], split into bf16 hi/lo so WMMA keeps ~16-bit mantissa.
    {
        const float4 h4 = *(const float4*)(hidden + (size_t)b * D_DIM + tid * 4);
        float hv[4] = {h4.x, h4.y, h4.z, h4.w};
        #pragma unroll
        for (int t = 0; t < 4; ++t) {
            __bf16 hi = (__bf16)hv[t];
            s_hh[tid * 4 + t] = hi;
            s_hl[tid * 4 + t] = (__bf16)(hv[t] - (float)hi);
        }
    }

    // Online-softmax running state; each lane owns d = jj*128 + lane*4 + t.
    float m_run = -INFINITY;
    float s_run = 0.0f;
    float c_run[16];
    #pragma unroll
    for (int i = 0; i < 16; ++i) c_run[i] = 0.0f;

    const size_t encBase = ((size_t)b * L_DIM + (size_t)chunk * CH_ROWS) * D_DIM;
    const int r0 = wave * 16;                      // this wave's 16-row slab

    for (int t = 0; t < N_TILE; ++t) {
        __syncthreads();   // s_enc reuse fence (also covers hidden staging, t==0)

        // Coalesced cooperative stage: 64 rows x 512 f32 -> padded LDS tile.
        {
            const float* gsrc = enc + encBase + (size_t)t * T_ROWS * D_DIM;
            #pragma unroll
            for (int k = 0; k < (T_ROWS * D_DIM / 4 / 128); ++k) {   // 64 iters
                int i   = tid + k * 128;      // float4 index in tile
                int row = i >> 7;             // 128 float4 per row
                int c4  = i & 127;
                float4 v = *(const float4*)(gsrc + (size_t)row * D_DIM + c4 * 4);
                *(float4*)(&s_enc[row * ROWPAD + c4 * 4]) = v;
            }
        }

        // Prefetch the next 128KB tile into WGP$/L2 (global_prefetch_b8) so
        // its HBM latency overlaps this tile's WMMA/VALU phase.
        if (t + 1 < N_TILE) {
            const char* nsrc =
                (const char*)(enc + encBase + (size_t)(t + 1) * T_ROWS * D_DIM);
            #pragma unroll
            for (int k = 0; k < 8; ++k)        // 128 thr * 8 * 128B = 128KB
                __builtin_prefetch(nsrc + ((size_t)tid + 128 * k) * 128, 0, 1);
        }
        __syncthreads();

        // ---- Energy for 16 rows via WMMA: C[m][*] = <enc_row_m, hidden> ----
        // A (16x32 bf16): lane = row m; half 0 carries K 0-15; half 1: 16-31.
        // B (32x16 bf16): hidden replicated over N; half 0 carries K 0-15; half 1: 16-31.
        const float* arow   = &s_enc[(r0 + mrow) * ROWPAD];
    const int    kbaseA = half * 16;
        const int    kbaseB = half * 16;
        v8f cacc = {};
        for (int kp = 0; kp < D_DIM; kp += 32) {
            float f[16];
            {
                const float* p0 = arow + kp + kbaseA;
                const float4 q0 = *(const float4*)(p0);
                const float4 q1 = *(const float4*)(p0 + 4);
                const float4 q2 = *(const float4*)(p0 + 8);
                const float4 q3 = *(const float4*)(p0 + 12);
                f[0]=q0.x;  f[1]=q0.y;  f[2]=q0.z;  f[3]=q0.w;
                f[4]=q1.x;  f[5]=q1.y;  f[6]=q1.z;  f[7]=q1.w;
                f[8]=q2.x;  f[9]=q2.y;  f[10]=q2.z; f[11]=q2.w;
                f[12]=q3.x; f[13]=q3.y; f[14]=q3.z; f[15]=q3.w;
            }
            v16bf ah, al;
            #pragma unroll
            for (int e = 0; e < 16; ++e) {
                __bf16 hi = (__bf16)f[e];
                ah[e] = hi;
                al[e] = (__bf16)(f[e] - (float)hi);
            }
            v16bf bh = *(const v16bf*)(&s_hh[kp + kbaseB]);
            v16bf bl = *(const v16bf*)(&s_hl[kp + kbaseB]);
            // e = Ah*Bh + Ah*Bl + Al*Bh  (drops only the ~2^-18 Al*Bl term)
            cacc = __builtin_amdgcn_wmma_f32_16x16x32_bf16(false, ah, false, bh,
                                                           (short)0, cacc, false, false);
            cacc = __builtin_amdgcn_wmma_f32_16x16x32_bf16(false, ah, false, bl,
                                                           (short)0, cacc, false, false);
            cacc = __builtin_amdgcn_wmma_f32_16x16x32_bf16(false, al, false, bh,
                                                           (short)0, cacc, false, false);
        }
        // Lane now holds energies for rows (half*8 + r) in cacc[r], r=0..7
        // (replicated across the 16 N columns of its half).

        // ---- Online softmax update over these 16 energies ----
        float lmax = cacc[0];
        #pragma unroll
        for (int r = 1; r < 8; ++r) lmax = fmaxf(lmax, cacc[r]);
        lmax = fmaxf(lmax, __shfl_xor(lmax, 16));
        const float new_m = fmaxf(m_run, lmax);
        const float alpha = __expf(m_run - new_m);   // 0 on first tile (m_run=-inf)
        float w[8];
        float lsum = 0.0f;
        #pragma unroll
        for (int r = 0; r < 8; ++r) { w[r] = __expf(cacc[r] - new_m); lsum += w[r]; }
        lsum += __shfl_xor(lsum, 16);
        s_run = s_run * alpha + lsum;
        m_run = new_m;

        // Gather weights for all 16 rows into every lane (constant indices).
        float wlo[8], whi[8];   // rows 0-7 / rows 8-15 of the slab
        #pragma unroll
        for (int r = 0; r < 8; ++r) {
            float mine  = w[r];
            float other = __shfl_xor(w[r], 16);
            wlo[r] = half ? other : mine;
            whi[r] = half ? mine  : other;
        }

        // ---- Rank-16 context accumulation from the LDS tile ----
        #pragma unroll
        for (int i = 0; i < 16; ++i) c_run[i] *= alpha;
        #pragma unroll
        for (int m = 0; m < 16; ++m) {
            const float wm = (m < 8) ? wlo[m] : whi[m - 8];
            const float* rp = &s_enc[(r0 + m) * ROWPAD];
            #pragma unroll
            for (int jj = 0; jj < 4; ++jj) {
                float4 v = *(const float4*)(rp + jj * 128 + lane * 4);
                c_run[jj * 4 + 0] = fmaf(wm, v.x, c_run[jj * 4 + 0]);
                c_run[jj * 4 + 1] = fmaf(wm, v.y, c_run[jj * 4 + 1]);
                c_run[jj * 4 + 2] = fmaf(wm, v.z, c_run[jj * 4 + 2]);
                c_run[jj * 4 + 3] = fmaf(wm, v.w, c_run[jj * 4 + 3]);
            }
        }
    }

    // ---- Merge the 4 wave partials (reuse s_enc) and emit block partial ----
    __syncthreads();
    {
        float* wp = &s_enc[wave * PARTSZ];
        #pragma unroll
        for (int jj = 0; jj < 4; ++jj) {
            float4 v = make_float4(c_run[jj * 4 + 0], c_run[jj * 4 + 1],
                                   c_run[jj * 4 + 2], c_run[jj * 4 + 3]);
            *(float4*)(wp + jj * 128 + lane * 4) = v;
        }
        if (lane == 0) { wp[512] = m_run; wp[513] = s_run; }
    }
    __syncthreads();
    {
        float mw0 = s_enc[0 * PARTSZ + 512], sw0 = s_enc[0 * PARTSZ + 513];
        float mw1 = s_enc[1 * PARTSZ + 512], sw1 = s_enc[1 * PARTSZ + 513];
        float mw2 = s_enc[2 * PARTSZ + 512], sw2 = s_enc[2 * PARTSZ + 513];
        float mw3 = s_enc[3 * PARTSZ + 512], sw3 = s_enc[3 * PARTSZ + 513];
        float M = fmaxf(fmaxf(mw0, mw1), fmaxf(mw2, mw3));
        float e0 = __expf(mw0 - M), e1 = __expf(mw1 - M);
        float e2 = __expf(mw2 - M), e3 = __expf(mw3 - M);
        float S  = sw0 * e0 + sw1 * e1 + sw2 * e2 + sw3 * e3;

        float* op = part + (size_t)(b * NCHUNK + chunk) * PARTSZ;
        #pragma unroll
        for (int t2 = 0; t2 < 4; ++t2) {
            int d = tid * 4 + t2;
            float acc = s_enc[0 * PARTSZ + d] * e0 + s_enc[1 * PARTSZ + d] * e1
                      + s_enc[2 * PARTSZ + d] * e2 + s_enc[3 * PARTSZ + d] * e3;
            op[d] = acc;
        }
        if (tid == 0) { op[512] = M; op[513] = S; }
    }
}

// Combine the 16 chunk-partials per batch (stable log-sum-exp) and apply /L.
__global__ __launch_bounds__(128) void attn_pass2(
    const float* __restrict__ part, float* __restrict__ out)
{
    const int b   = blockIdx.x;
    const int tid = threadIdx.x;
    __shared__ float sm[NCHUNK], ss[NCHUNK];
    if (tid < NCHUNK) {
        sm[tid] = part[(size_t)(b * NCHUNK + tid) * PARTSZ + 512];
        ss[tid] = part[(size_t)(b * NCHUNK + tid) * PARTSZ + 513];
    }
    __syncthreads();

    float G = -INFINITY;
    #pragma unroll
    for (int p = 0; p < NCHUNK; ++p) G = fmaxf(G, sm[p]);
    float sc[NCHUNK];
    float denom = 0.0f;
    #pragma unroll
    for (int p = 0; p < NCHUNK; ++p) { sc[p] = __expf(sm[p] - G); denom += ss[p] * sc[p]; }
    const float inv = 1.0f / (denom * (float)L_DIM);

    #pragma unroll
    for (int t2 = 0; t2 < 4; ++t2) {
        int d = tid * 4 + t2;
        float acc = 0.0f;
        #pragma unroll
        for (int p = 0; p < NCHUNK; ++p)
            acc += part[(size_t)(b * NCHUNK + p) * PARTSZ + d] * sc[p];
        out[(size_t)b * D_DIM + d] = acc * inv;
    }
}

extern "C" void kernel_launch(void* const* d_in, const int* in_sizes, int n_in,
                              void* d_out, int out_size, void* d_ws, size_t ws_size,
                              hipStream_t stream) {
    const float* hidden = (const float*)d_in[0];   // [64, 512]
    const float* enc    = (const float*)d_in[1];   // [64, 4096, 512]
    float* part = (float*)d_ws;                    // needs 64*16*514*4 ≈ 2.1 MB

    dim3 grid1(NCHUNK, B_DIM);
    attn_pass1<<<grid1, 128, 0, stream>>>(hidden, enc, part);
    attn_pass2<<<B_DIM, 128, 0, stream>>>(part, (float*)d_out);
}
